// MSAColumnAttention_40501541601318
// MI455X (gfx1250) — compile-verified, matrix operand order checked by reading, not run on hardware
//
#include <hip/hip_runtime.h>

#define SEQN 256   // N: MSA depth (attention axis)
#define LRES 384   // L: residues
#define CMD  256   // CM: model dim
#define NH   8
#define CHD  32
#define NTOK (SEQN * LRES)   // 98304 tokens
#define HC   (NH * CHD)      // 256

typedef __attribute__((ext_vector_type(16))) __bf16 v16bf;
typedef __attribute__((ext_vector_type(8)))  __bf16 v8bf;
typedef __attribute__((ext_vector_type(8)))  float  v8f;
typedef __attribute__((ext_vector_type(4)))  float  v4f;

// Load a 16x32 bf16 WMMA A-tile (or a 32x16 B-tile from transposed storage).
// ISA layout (CDNA5 ISA): lanes 0-15 hold M=lane, K {0..7,16..23};
// lanes 16-31 hold M=lane-16, K {8..15,24..31}.
static __device__ inline v16bf load_tile_bf16(const __bf16* __restrict__ base,
                                              int ld, int lane) {
  const int r  = lane & 15;
  const int kh = (lane >> 4) << 3;   // 0 or 8
  const __bf16* p = base + (size_t)r * ld + kh;
  v8bf lo = *(const v8bf*)(p);
  v8bf hi = *(const v8bf*)(p + 16);
  v16bf t;
#pragma unroll
  for (int i = 0; i < 8; ++i) { t[i] = lo[i]; t[i + 8] = hi[i]; }
  return t;
}

static __device__ inline v8f wmma_bf16(v16bf a, v16bf b, v8f c) {
  return __builtin_amdgcn_wmma_f32_16x16x32_bf16(
      /*neg_a=*/false, a, /*neg_b=*/false, b,
      /*c_mod=*/(short)0, c, /*reuse_a=*/false, /*reuse_b=*/false);
}

// ---------------------------------------------------------------------------
// Kernel 0: weight prep. WT4[1024][256] bf16 = [WqT;WkT;WvT;WgT],
// WoT[256][256] bf16 = Wo transposed. B-matrix columns become contiguous rows.
// ---------------------------------------------------------------------------
__global__ __launch_bounds__(256) void k_prep_weights(
    const float* __restrict__ Wq, const float* __restrict__ Wk,
    const float* __restrict__ Wv, const float* __restrict__ Wg,
    const float* __restrict__ Wo, __bf16* __restrict__ WT4,
    __bf16* __restrict__ WoT) {
  int idx = blockIdx.x * blockDim.x + threadIdx.x;
  if (idx < 4 * HC * CMD) {
    int cg = idx >> 8;       // output column 0..1023
    int kk = idx & 255;      // k
    int sec = cg >> 8;       // 0=q 1=k 2=v 3=g
    int col = cg & 255;
    const float* W = (sec == 0) ? Wq : (sec == 1) ? Wk : (sec == 2) ? Wv : Wg;
    WT4[idx] = (__bf16)W[kk * HC + col];
  } else {
    int j = idx - 4 * HC * CMD;
    if (j < CMD * HC) {
      int cm = j >> 8;       // output column of Wo
      int kk = j & 255;      // hc index
      WoT[j] = (__bf16)Wo[kk * CMD + cm];
    }
  }
}

// ---------------------------------------------------------------------------
// Kernel 1: LayerNorm, one wave per token, bf16 output.
// ---------------------------------------------------------------------------
__global__ __launch_bounds__(256) void k_layernorm(
    const float* __restrict__ m, const float* __restrict__ sc,
    const float* __restrict__ bi, __bf16* __restrict__ mn) {
  const int lane = threadIdx.x & 31;
  const int wave = threadIdx.x >> 5;
  const int tok = blockIdx.x * 8 + wave;
  const float* row = m + (size_t)tok * CMD;
  v4f x0 = *(const v4f*)(row + lane * 8);
  v4f x1 = *(const v4f*)(row + lane * 8 + 4);
  float s = 0.f;
#pragma unroll
  for (int i = 0; i < 4; ++i) s += x0[i] + x1[i];
#pragma unroll
  for (int off = 16; off; off >>= 1) s += __shfl_xor(s, off, 32);
  float mu = s * (1.f / CMD);
  float v = 0.f;
#pragma unroll
  for (int i = 0; i < 4; ++i) {
    float d0 = x0[i] - mu, d1 = x1[i] - mu;
    v += d0 * d0 + d1 * d1;
  }
#pragma unroll
  for (int off = 16; off; off >>= 1) v += __shfl_xor(v, off, 32);
  float rstd = rsqrtf(v * (1.f / CMD) + 1e-5f);
  v8bf yv;
#pragma unroll
  for (int i = 0; i < 4; ++i) {
    int c0 = lane * 8 + i, c1 = lane * 8 + 4 + i;
    yv[i]     = (__bf16)((x0[i] - mu) * rstd * sc[c0] + bi[c0]);
    yv[4 + i] = (__bf16)((x1[i] - mu) * rstd * sc[c1] + bi[c1]);
  }
  *(v8bf*)(mn + (size_t)tok * CMD + lane * 8) = yv;
}

// ---------------------------------------------------------------------------
// Kernel 2: fused QKVG projection.  [NTOK,256] @ [256,1024].
// Wave = 16 rows x 64 cols (A tile reused over 4 B tiles), K-loop 8x32.
// Q,K stored per-(l,h) row-major [n][ch]; V stored per-(l,h) transposed
// [ch][n] so attention loads it straight as a B matrix; gate pre-act bf16.
// ---------------------------------------------------------------------------
__global__ __launch_bounds__(256) void k_proj(
    const __bf16* __restrict__ mn, const __bf16* __restrict__ WT4,
    __bf16* __restrict__ qb, __bf16* __restrict__ kb,
    __bf16* __restrict__ vt, __bf16* __restrict__ gb) {
  const int lane = threadIdx.x & 31;
  const int wid = blockIdx.x * 8 + (threadIdx.x >> 5);
  const int rowTile = wid >> 4;       // 0..6143
  const int colGrp = wid & 15;        // 0..15, 64 cols each
  const int row0 = rowTile * 16;
  v8f acc[4] = {};
  for (int k0 = 0; k0 < CMD; k0 += 32) {
    v16bf a = load_tile_bf16(mn + (size_t)row0 * CMD + k0, CMD, lane);
#pragma unroll
    for (int j = 0; j < 4; ++j) {
      int cg0 = colGrp * 64 + j * 16;
      v16bf b = load_tile_bf16(WT4 + (size_t)cg0 * CMD + k0, CMD, lane);
      acc[j] = wmma_bf16(a, b, acc[j]);
    }
  }
  const int rbase = (lane >> 4) << 3;
  const int cl = lane & 15;
#pragma unroll
  for (int j = 0; j < 4; ++j) {
    int cg0 = colGrp * 64 + j * 16;
    int sec = cg0 >> 8;        // 0=q 1=k 2=v 3=g
    int cc = cg0 & 255;
    int h = cc >> 5;
    int c = (cc & 31) + cl;    // channel within head, stays in [0,32)
#pragma unroll
    for (int r = 0; r < 8; ++r) {
      int tok = row0 + rbase + r;
      int n = tok / LRES, l = tok - n * LRES;
      size_t lh = (size_t)l * NH + h;
      float val = acc[j][r];
      if (sec == 0)      qb[(lh * SEQN + n) * CHD + c] = (__bf16)val;
      else if (sec == 1) kb[(lh * SEQN + n) * CHD + c] = (__bf16)val;
      else if (sec == 2) vt[(lh * CHD + c) * SEQN + n] = (__bf16)val;
      else               gb[(size_t)tok * HC + cc + cl] = (__bf16)val;
    }
  }
}

// ---------------------------------------------------------------------------
// Kernel 3: attention per (l,h). Block = 4 waves; each wave owns 4 strips of
// 16 query rows. Logits (16x256) live in registers; softmax via shfl_xor row
// reductions (rows of a D-tile span lanes 0-15 / 16-31); bf16 P staged via
// LDS to re-layout into WMMA A form; P@V accumulated in 2 D tiles.
// ---------------------------------------------------------------------------
__global__ __launch_bounds__(128) void k_attn(
    const __bf16* __restrict__ qb, const __bf16* __restrict__ kb,
    const __bf16* __restrict__ vt, const float* __restrict__ msk,
    float* __restrict__ ao) {
  __shared__ __bf16 Pl[4][16][SEQN];   // 32 KB
  const int lane = threadIdx.x & 31;
  const int wave = threadIdx.x >> 5;
  const int lh = blockIdx.x;
  const int l = lh / NH, h = lh - (lh / NH) * NH;
  const __bf16* Q = qb + (size_t)lh * SEQN * CHD;
  const __bf16* K = kb + (size_t)lh * SEQN * CHD;
  const __bf16* V = vt + (size_t)lh * CHD * SEQN;
  const int rbase = (lane >> 4) << 3;
  const int cl = lane & 15;
  const int kh = rbase;               // A-tile K-half offset == (lane>>4)*8
  for (int s = wave; s < 16; s += 4) {
    const int n0 = s * 16;
    v16bf aq = load_tile_bf16(Q + (size_t)n0 * CHD, CHD, lane);
    v8f sc[16];
#pragma unroll
    for (int t = 0; t < 16; ++t) {
      v16bf bk = load_tile_bf16(K + (size_t)(t * 16) * CHD, CHD, lane);
      v8f z = {};
      z = wmma_bf16(aq, bk, z);
      float mv = msk[(size_t)(t * 16 + cl) * LRES + l];
#pragma unroll
      for (int r = 0; r < 8; ++r)
        sc[t][r] = (mv == 0.f) ? -1e9f : z[r] * 0.17677669529663688f;
    }
    // softmax: row (rbase+r) is spread over one 16-lane half
#pragma unroll
    for (int r = 0; r < 8; ++r) {
      float mx = -3.4e38f;
#pragma unroll
      for (int t = 0; t < 16; ++t) mx = fmaxf(mx, sc[t][r]);
#pragma unroll
      for (int off = 8; off; off >>= 1) mx = fmaxf(mx, __shfl_xor(mx, off, 32));
      float su = 0.f;
#pragma unroll
      for (int t = 0; t < 16; ++t) {
        float p = __expf(sc[t][r] - mx);
        sc[t][r] = p;
        su += p;
      }
#pragma unroll
      for (int off = 8; off; off >>= 1) su += __shfl_xor(su, off, 32);
      float ri = 1.f / su;
#pragma unroll
      for (int t = 0; t < 16; ++t)
        Pl[wave][rbase + r][t * 16 + cl] = (__bf16)(sc[t][r] * ri);
    }
    // P @ V : out 16x32, K-loop over m (8 steps of 32)
    v8f co0 = {}, co1 = {};
    for (int mk = 0; mk < 8; ++mk) {
      v16bf ap;
#pragma unroll
      for (int i = 0; i < 8; ++i) {
        ap[i]     = Pl[wave][cl][mk * 32 + kh + i];
        ap[i + 8] = Pl[wave][cl][mk * 32 + kh + 16 + i];
      }
      v16bf b0 = load_tile_bf16(V + (size_t)0  * SEQN + mk * 32, SEQN, lane);
      v16bf b1 = load_tile_bf16(V + (size_t)16 * SEQN + mk * 32, SEQN, lane);
      co0 = wmma_bf16(ap, b0, co0);
      co1 = wmma_bf16(ap, b1, co1);
    }
#pragma unroll
    for (int r = 0; r < 8; ++r) {
      int n = n0 + rbase + r;
      size_t tok = (size_t)n * LRES + l;
      ao[tok * HC + h * CHD + cl]      = co0[r];
      ao[tok * HC + h * CHD + 16 + cl] = co1[r];
    }
  }
}

// ---------------------------------------------------------------------------
// Kernel 4: gated output projection.  out = (ao * sigmoid(g+bg)) @ Wo + bo,
// masked. Wave = 16 rows x 64 cols, gated A-tile built on the fly.
// ---------------------------------------------------------------------------
__global__ __launch_bounds__(256) void k_out(
    const float* __restrict__ ao, const __bf16* __restrict__ gb,
    const float* __restrict__ bg, const __bf16* __restrict__ WoT,
    const float* __restrict__ bo, const float* __restrict__ msk,
    float* __restrict__ out) {
  const int lane = threadIdx.x & 31;
  const int wid = blockIdx.x * 8 + (threadIdx.x >> 5);
  const int rowTile = wid >> 2;       // 0..6143
  const int colGrp = wid & 3;         // 64 cols each
  const int row0 = rowTile * 16;
  const int rr = lane & 15;
  const int kh = (lane >> 4) << 3;
  v8f acc[4] = {};
  for (int k0 = 0; k0 < HC; k0 += 32) {
    const float*  arow = ao + (size_t)(row0 + rr) * HC + k0 + kh;
    const __bf16* grow = gb + (size_t)(row0 + rr) * HC + k0 + kh;
    v16bf a;
#pragma unroll
    for (int half = 0; half < 2; ++half) {
      int o = half * 16;
      v4f x0 = *(const v4f*)(arow + o);
      v4f x1 = *(const v4f*)(arow + o + 4);
      v8bf gg = *(const v8bf*)(grow + o);
#pragma unroll
      for (int i = 0; i < 8; ++i) {
        float gv = (float)gg[i] + bg[k0 + kh + o + i];
        float gate = 1.f / (1.f + __expf(-gv));
        float xv = (i < 4) ? x0[i] : x1[i - 4];
        a[half * 8 + i] = (__bf16)(xv * gate);
      }
    }
#pragma unroll
    for (int j = 0; j < 4; ++j) {
      v16bf b = load_tile_bf16(WoT + (size_t)(colGrp * 64 + j * 16) * HC + k0,
                               HC, lane);
      acc[j] = wmma_bf16(a, b, acc[j]);
    }
  }
  const int rbase = (lane >> 4) << 3;
  const int cl = lane & 15;
#pragma unroll
  for (int j = 0; j < 4; ++j) {
    int cm = colGrp * 64 + j * 16 + cl;
#pragma unroll
    for (int r = 0; r < 8; ++r) {
      int trow = row0 + rbase + r;
      out[(size_t)trow * CMD + cm] = (acc[j][r] + bo[cm]) * msk[trow];
    }
  }
}

// ---------------------------------------------------------------------------
extern "C" void kernel_launch(void* const* d_in, const int* in_sizes, int n_in,
                              void* d_out, int out_size, void* d_ws,
                              size_t ws_size, hipStream_t stream) {
  const float* m    = (const float*)d_in[0];
  const float* msk  = (const float*)d_in[1];   // [B,N,L] == [NTOK]
  const float* lsc  = (const float*)d_in[2];
  const float* lbi  = (const float*)d_in[3];
  const float* Wq   = (const float*)d_in[4];
  const float* Wk   = (const float*)d_in[5];
  const float* Wv   = (const float*)d_in[6];
  const float* Wg   = (const float*)d_in[7];
  const float* bg   = (const float*)d_in[8];
  const float* Wo   = (const float*)d_in[9];
  const float* bo   = (const float*)d_in[10];
  float* out = (float*)d_out;

  char* ws = (char*)d_ws;
  size_t off = 0;
  auto alloc = [&](size_t bytes) {
    char* p = ws + off;
    off += (bytes + 255) & ~(size_t)255;
    return p;
  };
  __bf16* mn  = (__bf16*)alloc((size_t)NTOK * CMD * 2);
  __bf16* WT4 = (__bf16*)alloc((size_t)4 * HC * CMD * 2);
  __bf16* WoT = (__bf16*)alloc((size_t)CMD * HC * 2);
  __bf16* qb  = (__bf16*)alloc((size_t)LRES * NH * SEQN * CHD * 2);
  __bf16* kb  = (__bf16*)alloc((size_t)LRES * NH * SEQN * CHD * 2);
  __bf16* vt  = (__bf16*)alloc((size_t)LRES * NH * CHD * SEQN * 2);
  __bf16* gb  = (__bf16*)alloc((size_t)NTOK * HC * 2);
  float*  ao  = (float*)alloc((size_t)NTOK * HC * 4);
  (void)ws_size;

  k_prep_weights<<<1280, 256, 0, stream>>>(Wq, Wk, Wv, Wg, Wo, WT4, WoT);
  k_layernorm<<<NTOK / 8, 256, 0, stream>>>(m, lsc, lbi, mn);
  k_proj<<<(NTOK / 16) * 16 / 8, 256, 0, stream>>>(mn, WT4, qb, kb, vt, gb);
  k_attn<<<LRES * NH, 128, 0, stream>>>(qb, kb, vt, msk, ao);
  k_out<<<(NTOK / 16) * 4 / 8, 256, 0, stream>>>(ao, gb, bg, WoT, bo, msk, out);
}
